// GatedDeltaMixer_7103875907803
// MI455X (gfx1250) — matrix operand order repaired and checked by reading, not compiled, not run on hardware
//
#include <hip/hip_runtime.h>
#include <hip/hip_bf16.h>

#define BD 8
#define NS 2048
#define CD 256
#define MROWS (BD * NS)   // 16384
#define TBLK 8            // timesteps staged per TDM transfer

typedef __attribute__((ext_vector_type(16))) __bf16 v16bf;
typedef __attribute__((ext_vector_type(8)))  float  v8f;
typedef unsigned int v4ui __attribute__((ext_vector_type(4)));
typedef int          v8si __attribute__((ext_vector_type(8)));
typedef int          v4si __attribute__((ext_vector_type(4)));

// ---------------- fp32 -> bf16 convert ----------------
__global__ void cvt_f32_bf16(const float* __restrict__ in, __bf16* __restrict__ out, int n) {
    int i = blockIdx.x * blockDim.x + threadIdx.x;
    if (i < n) out[i] = (__bf16)in[i];
}

// ---------------- WMMA fragment loaders (bf16, 16x16x32 shapes) ----------------
// A 16x32: lanes 0-15: VGPR0-3 -> K0..7, VGPR4-7 -> K16..23; lanes16-31: K8..15 / K24..31
__device__ __forceinline__ v16bf load_fragA(const __bf16* __restrict__ ap, int kb, int half) {
    v16bf f;
#pragma unroll
    for (int i = 0; i < 8; ++i) {
        int ka = (i < 4) ? (2 * i + 8 * half) : (16 + 2 * (i - 4) + 8 * half);
        f[2 * i]     = ap[kb + ka];
        f[2 * i + 1] = ap[kb + ka + 1];
    }
    return f;
}
// B 32x16: lanes 0-15: VGPR0-3 -> K0..7, VGPR4-7 -> K16..23; lanes16-31: K8..15 / K24..31
__device__ __forceinline__ v16bf load_fragB(const __bf16* __restrict__ bp, int kb, int half) {
    v16bf f;
#pragma unroll
    for (int i = 0; i < 8; ++i) {
        int kk = (i < 4) ? (2 * i + 8 * half) : (16 + 2 * (i - 4) + 8 * half);
        f[2 * i]     = bp[kb + kk];
        f[2 * i + 1] = bp[kb + kk + 1];
    }
    return f;
}

// ---------------- WMMA GEMM: Y[M,N] = act(A[M,K] @ W[N,K]^T + bias[N]) ----------------
// act: 0 = none, 1 = silu. One wave32 computes a 16x64 tile (4 accumulators),
// reusing the A fragment 4x per K-step -> 4 v_wmma per 5 fragment loads.
__global__ void gemm_wmma_bf16(const __bf16* __restrict__ A,
                               const __bf16* __restrict__ W,
                               const float*  __restrict__ bias,
                               float*        __restrict__ Y,
                               int M, int N, int K, int act)
{
    const int lane   = threadIdx.x & 31;
    const int wave   = threadIdx.x >> 5;
    const int tilesN = N >> 6;                       // 64-wide column groups
    const int tile   = blockIdx.x * (blockDim.x >> 5) + wave;
    const int m0 = (tile / tilesN) << 4;
    const int n0 = (tile % tilesN) << 6;
    const int half = lane >> 4;
    const int r    = lane & 15;

    v8f c0 = {}, c1 = {}, c2 = {}, c3 = {};
    const __bf16* ap  = A + (size_t)(m0 + r) * K;
    const __bf16* bp0 = W + (size_t)(n0 +  0 + r) * K;   // B[k][n] = W[n][k]
    const __bf16* bp1 = W + (size_t)(n0 + 16 + r) * K;
    const __bf16* bp2 = W + (size_t)(n0 + 32 + r) * K;
    const __bf16* bp3 = W + (size_t)(n0 + 48 + r) * K;

    for (int kb = 0; kb < K; kb += 32) {
        v16bf af = load_fragA(ap, kb, half);
        v16bf b0 = load_fragB(bp0, kb, half);
        v16bf b1 = load_fragB(bp1, kb, half);
        v16bf b2 = load_fragB(bp2, kb, half);
        v16bf b3 = load_fragB(bp3, kb, half);
        c0 = __builtin_amdgcn_wmma_f32_16x16x32_bf16(false, af, false, b0, (short)0, c0, false, false);
        c1 = __builtin_amdgcn_wmma_f32_16x16x32_bf16(false, af, false, b1, (short)0, c1, false, false);
        c2 = __builtin_amdgcn_wmma_f32_16x16x32_bf16(false, af, false, b2, (short)0, c2, false, false);
        c3 = __builtin_amdgcn_wmma_f32_16x16x32_bf16(false, af, false, b3, (short)0, c3, false, false);
    }

#pragma unroll
    for (int j = 0; j < 4; ++j) {
        const v8f  c    = (j == 0) ? c0 : (j == 1) ? c1 : (j == 2) ? c2 : c3;
        const int  nOut = n0 + 16 * j + r;
        const float bn  = bias[nOut];
#pragma unroll
        for (int vr = 0; vr < 8; ++vr) {
            int   mOut = m0 + vr + 8 * half;     // D: VGPR vr -> M=vr / vr+8
            float val  = c[vr] + bn;
            if (act == 1) val = val / (1.0f + __expf(-val));   // silu
            Y[(size_t)mOut * N + nOut] = val;
        }
    }
}

// ---------------- row-wise L2 normalize (one wave32 per row) ----------------
__global__ void l2norm_rows(float* __restrict__ X, int C) {
    const int row  = blockIdx.x;
    const int lane = threadIdx.x;
    float* xr = X + (size_t)row * C;
    float s = 0.0f;
    for (int j = lane; j < C; j += 32) { float v = xr[j]; s += v * v; }
#pragma unroll
    for (int off = 16; off > 0; off >>= 1) s += __shfl_xor(s, off, 32);
    const float inv = 1.0f / (sqrtf(s) + 1e-6f);
    for (int j = lane; j < C; j += 32) xr[j] *= inv;
}

// ---------------- row-wise mean(sigmoid(.)) (one wave32 per row) ----------------
__global__ void sigmoid_mean_rows(const float* __restrict__ X, float* __restrict__ out, int C) {
    const int row  = blockIdx.x;
    const int lane = threadIdx.x;
    const float* xr = X + (size_t)row * C;
    float s = 0.0f;
    for (int j = lane; j < C; j += 32) s += 1.0f / (1.0f + __expf(-xr[j]));
#pragma unroll
    for (int off = 16; off > 0; off >>= 1) s += __shfl_xor(s, off, 32);
    if (lane == 0) out[row] = s / (float)C;
}

// ---------------- TDM: async 2-D tile load (Global -> LDS) ----------------
// Builds the D# per CDNA5 ISA (CDNA5 ISA Sect.8):
//  group0: count=1 | lds_addr | global_addr[56:0] | type=2
//  group1: data_size=4B | tensor_dim0/1 | tile_dim0/1 | tensor_dim0_stride
//  groups2/3(+aux): zero (2-D tile -> higher dims unused)
__device__ __forceinline__ void tdm_load_2d(const float* __restrict__ gsrc,
                                            unsigned int lds_byte,
                                            int tile_rows, int cols,
                                            int stride_elems, int tensor_rows)
{
    unsigned long long ga = (unsigned long long)(const void*)gsrc;
    v4ui g0;
    g0[0] = 1u;                                               // count=1, user mode
    g0[1] = lds_byte;                                         // lds_addr (bytes)
    g0[2] = (unsigned int)(ga & 0xFFFFFFFFull);               // global_addr[31:0]
    g0[3] = (unsigned int)((ga >> 32) & 0x1FFFFFFull)         // global_addr[56:32]
          | (2u << 30);                                       // type=2 ("image")
    v8si g1;
    g1[0] = (int)(2u << 16);                                  // data_size=2 -> 4 bytes
    g1[1] = (int)(((unsigned)cols & 0xFFFFu) << 16);          // tensor_dim0[15:0]
    g1[2] = (int)((((unsigned)cols >> 16) & 0xFFFFu)
          | (((unsigned)tensor_rows & 0xFFFFu) << 16));       // td0[31:16] | td1[15:0]
    g1[3] = (int)((((unsigned)tensor_rows >> 16) & 0xFFFFu)
          | (((unsigned)cols & 0xFFFFu) << 16));              // td1[31:16] | tile_dim0
    g1[4] = (int)((unsigned)tile_rows & 0xFFFFu);             // tile_dim1 (tile_dim2=0)
    g1[5] = stride_elems;                                     // tensor_dim0_stride[31:0]
    g1[6] = 0;
    g1[7] = 0;
    v4si g2  = {0, 0, 0, 0};
    v4si g3  = {0, 0, 0, 0};
    v8si gx  = {0, 0, 0, 0, 0, 0, 0, 0};
    // 6-arg form (clang-23 / therock-10.0 HIP headers)
    __builtin_amdgcn_tensor_load_to_lds(g0, g1, g2, g3, gx, 0);
}

// ---------------- gated delta scan: S in LDS, TDM double-buffered q/k/v ----------------
// S_t = a_t * S + (b_t*v - a_t*b_t*(S k)) k^T ;  o_t = S_t q
// Thread i owns row i of S. q/k/v for TBLK steps are staged by the Tensor Data
// Mover into LDS (wave 0 issues; s_wait_tensorcnt + one barrier per TBLK steps).
__global__ void delta_scan(const float* __restrict__ q,
                           const float* __restrict__ k,
                           const float* __restrict__ v,
                           const float* __restrict__ a,
                           const float* __restrict__ bg,
                           float*       __restrict__ O,
                           int N, int C)
{
    extern __shared__ float smem[];
    const int STR   = CD + 4;                  // 260 floats: 16B-aligned rows, bank skew
    const int SELEM = CD * STR;                // S footprint (floats)
    const int BUFE  = TBLK * CD;               // one array tile (floats)
    float* S = smem;                           // [C][STR]
    // staging: [2 buffers][q,k,v][TBLK*CD]
    float* stage = smem + SELEM;
    const unsigned int stageByte0 = (unsigned int)(SELEM * sizeof(float));

    const int b   = blockIdx.x;
    const int tid = threadIdx.x;               // row owner, 0..C-1
    float* Srow = S + tid * STR;
    for (int j = 0; j < C; j += 4) *(float4*)(Srow + j) = make_float4(0.f, 0.f, 0.f, 0.f);

    const size_t base = (size_t)b * N * C;
    const float* qb = q + base;
    const float* kb = k + base;
    const float* vb = v + base;

    // prologue: stage block 0 into buffer 0
    if (tid < 32) {
        tdm_load_2d(qb, stageByte0 + 0 * BUFE * 4, TBLK, CD, CD, N);
        tdm_load_2d(kb, stageByte0 + 1 * BUFE * 4, TBLK, CD, CD, N);
        tdm_load_2d(vb, stageByte0 + 2 * BUFE * 4, TBLK, CD, CD, N);
        __builtin_amdgcn_s_wait_tensorcnt((short)0);
    }
    __syncthreads();

    int cur = 0;
    for (int tb = 0; tb < N; tb += TBLK) {
        // prefetch next block into the other buffer while we compute
        if (tb + TBLK < N && tid < 32) {
            const unsigned int nb = stageByte0 + (unsigned int)((1 - cur) * 3 * BUFE) * 4u;
            const size_t goff = (size_t)(tb + TBLK) * C;
            tdm_load_2d(qb + goff, nb + 0u * BUFE * 4u, TBLK, CD, CD, N);
            tdm_load_2d(kb + goff, nb + 1u * BUFE * 4u, TBLK, CD, CD, N);
            tdm_load_2d(vb + goff, nb + 2u * BUFE * 4u, TBLK, CD, CD, N);
        }
        const float* qs = stage + cur * 3 * BUFE + 0 * BUFE;
        const float* ks = stage + cur * 3 * BUFE + 1 * BUFE;
        const float* vs = stage + cur * 3 * BUFE + 2 * BUFE;

        // TBLK dependent steps; no barriers needed (row ownership + read-only stage)
        for (int tt = 0; tt < TBLK; ++tt) {
            const int   t  = tb + tt;
            const float at = a[b * N + t];
            const float bt = bg[b * N + t];
            const float* kv = ks + tt * CD;
            const float* qv = qs + tt * CD;

            float u = 0.0f;                    // u_i = (S k)_i
            for (int j = 0; j < C; j += 4) {
                float4 s4 = *(const float4*)(Srow + j);
                float4 k4 = *(const float4*)(kv + j);
                u += s4.x * k4.x + s4.y * k4.y + s4.z * k4.z + s4.w * k4.w;
            }
            const float coef = bt * vs[tt * CD + tid] - at * bt * u;

            float o = 0.0f;                    // fused rank-1 update + o_i = (S_new q)_i
            for (int j = 0; j < C; j += 4) {
                float4 s4 = *(const float4*)(Srow + j);
                float4 k4 = *(const float4*)(kv + j);
                float4 q4 = *(const float4*)(qv + j);
                s4.x = at * s4.x + coef * k4.x;
                s4.y = at * s4.y + coef * k4.y;
                s4.z = at * s4.z + coef * k4.z;
                s4.w = at * s4.w + coef * k4.w;
                o += s4.x * q4.x + s4.y * q4.y + s4.z * q4.z + s4.w * q4.w;
                *(float4*)(Srow + j) = s4;
            }
            O[base + (size_t)t * C + tid] = o;
        }

        if (tid < 32) __builtin_amdgcn_s_wait_tensorcnt((short)0);
        __syncthreads();                       // next buffer ready; cur buffer reusable
        cur ^= 1;
    }
}

extern "C" void kernel_launch(void* const* d_in, const int* in_sizes, int n_in,
                              void* d_out, int out_size, void* d_ws, size_t ws_size,
                              hipStream_t stream) {
    const float* x  = (const float*)d_in[0];
    const float* Wq = (const float*)d_in[1];  const float* bq = (const float*)d_in[2];
    const float* Wk = (const float*)d_in[3];  const float* bk = (const float*)d_in[4];
    const float* Wv = (const float*)d_in[5];  const float* bv = (const float*)d_in[6];
    const float* Wa = (const float*)d_in[7];  const float* ba = (const float*)d_in[8];
    const float* Wb = (const float*)d_in[9];  const float* bb = (const float*)d_in[10];
    const float* Wo = (const float*)d_in[11]; const float* bo = (const float*)d_in[12];

    char*  ws  = (char*)d_ws;
    size_t off = 0;
    auto alloc = [&](size_t bytes) -> void* {
        void* p = ws + off;
        off += (bytes + 255) & ~(size_t)255;
        return p;
    };

    const int nx = MROWS * CD;              // 4,194,304
    const int nw = CD * CD;                 // 65,536

    __bf16* x_bf  = (__bf16*)alloc((size_t)nx * sizeof(__bf16));
    __bf16* Wq_bf = (__bf16*)alloc((size_t)nw * sizeof(__bf16));
    __bf16* Wk_bf = (__bf16*)alloc((size_t)nw * sizeof(__bf16));
    __bf16* Wv_bf = (__bf16*)alloc((size_t)nw * sizeof(__bf16));
    __bf16* Wa_bf = (__bf16*)alloc((size_t)nw * sizeof(__bf16));
    __bf16* Wb_bf = (__bf16*)alloc((size_t)nw * sizeof(__bf16));
    __bf16* Wo_bf = (__bf16*)alloc((size_t)nw * sizeof(__bf16));
    float*  qf    = (float*)alloc((size_t)nx * sizeof(float));
    float*  kf    = (float*)alloc((size_t)nx * sizeof(float));
    float*  vf    = (float*)alloc((size_t)nx * sizeof(float));
    float*  tmp   = (float*)alloc((size_t)nx * sizeof(float));   // a/b logits, then O
    float*  avec  = (float*)alloc((size_t)MROWS * sizeof(float));
    float*  bvec  = (float*)alloc((size_t)MROWS * sizeof(float));
    __bf16* o_bf  = (__bf16*)alloc((size_t)nx * sizeof(__bf16));

    // --- convert activations + weights to bf16 ---
    cvt_f32_bf16<<<(nx + 255) / 256, 256, 0, stream>>>(x,  x_bf,  nx);
    cvt_f32_bf16<<<(nw + 255) / 256, 256, 0, stream>>>(Wq, Wq_bf, nw);
    cvt_f32_bf16<<<(nw + 255) / 256, 256, 0, stream>>>(Wk, Wk_bf, nw);
    cvt_f32_bf16<<<(nw + 255) / 256, 256, 0, stream>>>(Wv, Wv_bf, nw);
    cvt_f32_bf16<<<(nw + 255) / 256, 256, 0, stream>>>(Wa, Wa_bf, nw);
    cvt_f32_bf16<<<(nw + 255) / 256, 256, 0, stream>>>(Wb, Wb_bf, nw);
    cvt_f32_bf16<<<(nw + 255) / 256, 256, 0, stream>>>(Wo, Wo_bf, nw);

    // --- projections via WMMA: 16x64 tile per wave, 8 waves per block ---
    const int gemmBlocks = (MROWS / 16) * (CD / 64) / 8;   // 512

    gemm_wmma_bf16<<<gemmBlocks, 256, 0, stream>>>(x_bf, Wq_bf, bq, qf, MROWS, CD, CD, 1);
    gemm_wmma_bf16<<<gemmBlocks, 256, 0, stream>>>(x_bf, Wk_bf, bk, kf, MROWS, CD, CD, 1);
    gemm_wmma_bf16<<<gemmBlocks, 256, 0, stream>>>(x_bf, Wv_bf, bv, vf, MROWS, CD, CD, 1);

    gemm_wmma_bf16<<<gemmBlocks, 256, 0, stream>>>(x_bf, Wa_bf, ba, tmp, MROWS, CD, CD, 0);
    sigmoid_mean_rows<<<MROWS, 32, 0, stream>>>(tmp, avec, CD);
    gemm_wmma_bf16<<<gemmBlocks, 256, 0, stream>>>(x_bf, Wb_bf, bb, tmp, MROWS, CD, CD, 0);
    sigmoid_mean_rows<<<MROWS, 32, 0, stream>>>(tmp, bvec, CD);

    l2norm_rows<<<MROWS, 32, 0, stream>>>(qf, CD);
    l2norm_rows<<<MROWS, 32, 0, stream>>>(kf, CD);

    // --- sequential gated-delta scan: S + double-buffered TDM staging in LDS ---
    const size_t smemBytes =
        ((size_t)(CD + 4) * CD + 2 * 3 * TBLK * CD) * sizeof(float);   // ~315 KB
    (void)hipFuncSetAttribute(reinterpret_cast<const void*>(delta_scan),
                              hipFuncAttributeMaxDynamicSharedMemorySize, (int)smemBytes);
    delta_scan<<<BD, CD, smemBytes, stream>>>(qf, kf, vf, avec, bvec, tmp, NS, CD);

    // --- output projection ---
    cvt_f32_bf16<<<(nx + 255) / 256, 256, 0, stream>>>(tmp, o_bf, nx);
    gemm_wmma_bf16<<<gemmBlocks, 256, 0, stream>>>(o_bf, Wo_bf, bo, (float*)d_out,
                                                   MROWS, CD, CD, 0);
}
